// DecisionMaker_23029614641600
// MI455X (gfx1250) — compile-verified, matrix operand order checked by reading, not run on hardware
//
#include <hip/hip_runtime.h>
#include <hip/hip_bf16.h>

// ---------------- problem constants ----------------
#define DIMD    256
#define BDIM    1024
#define NITEMS  100000
#define KTOP    100
#define EHIST   51200
#define MINIDX  1

typedef __attribute__((ext_vector_type(16))) __bf16 v16bf;
typedef __attribute__((ext_vector_type(8)))  float  v8f;

union BF16Frag { v16bf v; unsigned short u[16]; };

__device__ inline unsigned short f2bf(float f) {
  unsigned int u = __float_as_uint(f);
  unsigned int r = u + 0x7FFFu + ((u >> 16) & 1u);   // round-to-nearest-even
  return (unsigned short)(r >> 16);
}

__device__ inline v8f wmma_bf16(v16bf a, v16bf b, v8f c) {
  return __builtin_amdgcn_wmma_f32_16x16x32_bf16(false, a, false, b, (short)0, c,
                                                 false, false);
}

// A-frag (16x32 bf16) from an LDS tile stored row-major [16][256] as u16.
// Lane l: M = l&15; lanes 0-15 hold K={0..7,16..23}, lanes 16-31 K={8..15,24..31}.
__device__ inline v16bf afrag_lds(const unsigned short* s, int k0, int lane) {
  BF16Frag f;
  const unsigned short* row = s + (lane & 15) * DIMD + k0 + ((lane >> 4) * 8);
#pragma unroll
  for (int i = 0; i < 8; ++i) f.u[i] = row[i];
#pragma unroll
  for (int i = 0; i < 8; ++i) f.u[8 + i] = row[16 + i];
  return f.v;
}

// B-frag (32x16): logical B[k][n] = M[(row0+n)*256 + k] where M is a bf16
// matrix stored row-major with 256 columns (item matrix, or transposed weight).
// Each lane reads two contiguous 16-byte runs -> wide coalesced loads.
__device__ inline v16bf bfrag_rows(const unsigned short* M, int row0, int k0,
                                   int lane) {
  const unsigned short* p = M + (size_t)(row0 + (lane & 15)) * DIMD + k0 +
                            ((lane >> 4) * 8);
  BF16Frag f;
#pragma unroll
  for (int i = 0; i < 8; ++i) f.u[i] = p[i];
#pragma unroll
  for (int i = 0; i < 8; ++i) f.u[8 + i] = p[16 + i];
  return f.v;
}

// Mean / rsqrt(var+eps) of a 256-wide row shared by a 16-lane group.
__device__ inline void row_stats(const float* rowp, int l16, float* mean,
                                 float* rstd) {
  float s1 = 0.f, s2 = 0.f;
#pragma unroll
  for (int j = 0; j < 16; ++j) {
    float v = rowp[l16 + j * 16];
    s1 += v; s2 += v * v;
  }
#pragma unroll
  for (int off = 8; off >= 1; off >>= 1) {
    s1 += __shfl_xor(s1, off, 16);
    s2 += __shfl_xor(s2, off, 16);
  }
  float m = s1 * (1.f / 256.f);
  float var = s2 * (1.f / 256.f) - m * m;
  *mean = m;
  *rstd = rsqrtf(var + 1e-5f);
}

// ------------- one-time weight prep: W (f32 KxN) -> W^T (bf16 NxK) ----------
__global__ __launch_bounds__(256) void prep_wT_kernel(
    const float* __restrict__ W, unsigned short* __restrict__ WT) {
  const int n = threadIdx.x;        // coalesced read along N
  const int k = blockIdx.x;
  WT[(size_t)n * DIMD + k] = f2bf(W[(size_t)k * DIMD + n]);
}

// ------------- encode: LN(relu(LN(x@W1+b1))@W2+b2), bf16 output -------------
__global__ __launch_bounds__(256) void encode_kernel(
    const float* __restrict__ X,
    const unsigned short* __restrict__ W1T, const float* __restrict__ b1,
    const float* __restrict__ g1, const float* __restrict__ be1,
    const unsigned short* __restrict__ W2T, const float* __restrict__ b2,
    const float* __restrict__ g2, const float* __restrict__ be2,
    unsigned short* __restrict__ out) {
  __shared__ unsigned short sA[16 * DIMD];   // bf16 activations (GEMM input)
  __shared__ float sH[16 * DIMD];            // f32 GEMM result for LayerNorm

  const int tid  = threadIdx.x;
  const int lane = tid & 31;
  const int wave = tid >> 5;
  const int row0 = blockIdx.x * 16;
  const int n_in = lane & 15;
  const int mofs = (lane >> 4) * 8;
  const int lnrow = wave * 2 + (lane >> 4);
  const int l16   = lane & 15;

  // stage input tile as bf16
  for (int i = tid; i < 16 * DIMD; i += 256)
    sA[i] = f2bf(X[(size_t)(row0 + (i >> 8)) * DIMD + (i & 255)]);
  __syncthreads();

  // ---- layer 1 GEMM: h = x @ W1 ----
  {
    v8f acc0 = {}, acc1 = {};
    const int n0 = wave * 32;
#pragma unroll
    for (int kk = 0; kk < 8; ++kk) {
      v16bf a = afrag_lds(sA, kk * 32, lane);
      acc0 = wmma_bf16(a, bfrag_rows(W1T, n0,      kk * 32, lane), acc0);
      acc1 = wmma_bf16(a, bfrag_rows(W1T, n0 + 16, kk * 32, lane), acc1);
    }
#pragma unroll
    for (int v = 0; v < 8; ++v) {
      int m = mofs + v;
      sH[m * DIMD + n0 + n_in]      = acc0[v] + b1[n0 + n_in];
      sH[m * DIMD + n0 + 16 + n_in] = acc1[v] + b1[n0 + 16 + n_in];
    }
  }
  __syncthreads();

  // ---- LN1 + relu -> bf16 activations ----
  {
    const float* rowp = sH + lnrow * DIMD;
    float m, rs; row_stats(rowp, l16, &m, &rs);
#pragma unroll
    for (int j = 0; j < 16; ++j) {
      int c = l16 + j * 16;
      float y = (rowp[c] - m) * rs * g1[c] + be1[c];
      sA[lnrow * DIMD + c] = f2bf(fmaxf(y, 0.f));
    }
  }
  __syncthreads();

  // ---- layer 2 GEMM: h @ W2 ----
  {
    v8f acc0 = {}, acc1 = {};
    const int n0 = wave * 32;
#pragma unroll
    for (int kk = 0; kk < 8; ++kk) {
      v16bf a = afrag_lds(sA, kk * 32, lane);
      acc0 = wmma_bf16(a, bfrag_rows(W2T, n0,      kk * 32, lane), acc0);
      acc1 = wmma_bf16(a, bfrag_rows(W2T, n0 + 16, kk * 32, lane), acc1);
    }
#pragma unroll
    for (int v = 0; v < 8; ++v) {
      int m = mofs + v;
      sH[m * DIMD + n0 + n_in]      = acc0[v] + b2[n0 + n_in];
      sH[m * DIMD + n0 + 16 + n_in] = acc1[v] + b2[n0 + 16 + n_in];
    }
  }
  __syncthreads();

  // ---- LN2 -> bf16 global output ----
  {
    const float* rowp = sH + lnrow * DIMD;
    float m, rs; row_stats(rowp, l16, &m, &rs);
    unsigned short* orow = out + (size_t)(row0 + lnrow) * DIMD;
#pragma unroll
    for (int j = 0; j < 16; ++j) {
      int c = l16 + j * 16;
      orow[c] = f2bf((rowp[c] - m) * rs * g2[c] + be2[c]);
    }
  }
}

// ------------- scores = U @ IT^T  (16 users x 256 items per block) -----------
__global__ __launch_bounds__(256) void scores_kernel(
    const unsigned short* __restrict__ U, const unsigned short* __restrict__ IT,
    float* __restrict__ S) {
  __shared__ unsigned short sU[16 * DIMD];
  const int tid  = threadIdx.x;
  const int lane = tid & 31;
  const int wave = tid >> 5;
  const int urow0 = blockIdx.y * 16;
  const int col0  = blockIdx.x * 256;

  for (int i = tid; i < 16 * DIMD; i += 256)
    sU[i] = U[(size_t)(urow0 + (i >> 8)) * DIMD + (i & 255)];
  __syncthreads();

  const int nb0 = col0 + wave * 16;
  const int nb1 = col0 + (wave + 8) * 16;
  const bool ok0 = nb0 < NITEMS;   // wave-uniform (EXEC stays all-ones for WMMA)
  const bool ok1 = nb1 < NITEMS;

  v8f acc0 = {}, acc1 = {};
#pragma unroll
  for (int kk = 0; kk < 8; ++kk) {
    v16bf a = afrag_lds(sU, kk * 32, lane);
    if (ok0) acc0 = wmma_bf16(a, bfrag_rows(IT, nb0, kk * 32, lane), acc0);
    if (ok1) acc1 = wmma_bf16(a, bfrag_rows(IT, nb1, kk * 32, lane), acc1);
  }

  const int n_in = lane & 15;
  const int mofs = (lane >> 4) * 8;
  if (ok0) {
#pragma unroll
    for (int v = 0; v < 8; ++v)
      S[(size_t)(urow0 + mofs + v) * NITEMS + nb0 + n_in] = acc0[v];
  }
  if (ok1) {
#pragma unroll
    for (int v = 0; v < 8; ++v)
      S[(size_t)(urow0 + mofs + v) * NITEMS + nb1 + n_in] = acc1[v];
  }
}

// ------------- scatter zeros for history entries -----------------------------
__global__ void zero_hist_kernel(const int* __restrict__ rows,
                                 const int* __restrict__ cols,
                                 float* __restrict__ S, int e) {
  int i = blockIdx.x * blockDim.x + threadIdx.x;
  if (i < e) S[(size_t)rows[i] * NITEMS + cols[i]] = 0.f;
}

// ------------- probs = sigmoid(scores), float4 stream ------------------------
__global__ void sigmoid_kernel(const float* __restrict__ S,
                               float* __restrict__ P, long long n4) {
  long long i = (long long)blockIdx.x * blockDim.x + threadIdx.x;
  const long long stride = (long long)gridDim.x * blockDim.x;
  const float4* s4 = (const float4*)S;
  float4* p4 = (float4*)P;
  for (; i < n4; i += stride) {
    float4 v = s4[i], r;
    r.x = 1.f / (1.f + __expf(-v.x));
    r.y = 1.f / (1.f + __expf(-v.y));
    r.z = 1.f / (1.f + __expf(-v.z));
    r.w = 1.f / (1.f + __expf(-v.w));
    p4[i] = r;
  }
}

// ------------- per-row top-K via radix-select over 2048 bins -----------------
#define NBINS 2048
#define CAP   2048

__device__ inline unsigned int fkey(float f) {   // order-preserving uint key
  unsigned int u = __float_as_uint(f);
  return (u & 0x80000000u) ? ~u : (u | 0x80000000u);
}

__global__ __launch_bounds__(256) void topk_kernel(const float* __restrict__ S,
                                                   float* __restrict__ vals,
                                                   int* __restrict__ idx) {
  __shared__ unsigned int hist[NBINS];
  __shared__ float cv[CAP];
  __shared__ int   ci[CAP];
  __shared__ unsigned int cnt;
  __shared__ int thrBin;
  __shared__ float rv[256];
  __shared__ int   ri[256];

  const int tid = threadIdx.x;
  const int row = blockIdx.x;
  const float* srow = S + (size_t)row * NITEMS;

  for (int i = tid; i < NBINS; i += 256) hist[i] = 0u;
  if (tid == 0) cnt = 0u;
  __syncthreads();

  // pass 1: histogram of top-11-bit keys (LDS atomics)
  for (int i = tid; i < NITEMS; i += 256)
    atomicAdd(&hist[fkey(srow[i]) >> 21], 1u);
  __syncthreads();

  // threshold bin: largest bin b such that count(key-bin >= b) >= K
  if (tid == 0) {
    unsigned int cum = 0;
    int b = NBINS - 1;
    for (; b > 0; --b) { cum += hist[b]; if (cum >= KTOP) break; }
    thrBin = b;
  }
  __syncthreads();

  // pass 2: gather candidates
  const unsigned int tb = (unsigned int)thrBin;
  for (int i = tid; i < NITEMS; i += 256) {
    float v = srow[i];
    if ((fkey(v) >> 21) >= tb) {
      unsigned int slot = atomicAdd(&cnt, 1u);
      if (slot < CAP) { cv[slot] = v; ci[slot] = i; }
    }
  }
  __syncthreads();

  const int m = (cnt < (unsigned)CAP) ? (int)cnt : CAP;

  // pass 3: K max-extractions over the candidate pool
  for (int k = 0; k < KTOP; ++k) {
    float bv = -3.4e38f; int bi = -1;
    for (int j = tid; j < m; j += 256)
      if (cv[j] > bv) { bv = cv[j]; bi = j; }
    rv[tid] = bv; ri[tid] = bi;
    __syncthreads();
    for (int s = 128; s > 0; s >>= 1) {
      if (tid < s && rv[tid + s] > rv[tid]) {
        rv[tid] = rv[tid + s]; ri[tid] = ri[tid + s];
      }
      __syncthreads();
    }
    if (tid == 0) {
      int slot = ri[0];
      if (slot >= 0) {
        vals[(size_t)row * KTOP + k] = cv[slot];
        idx[(size_t)row * KTOP + k]  = ci[slot] + MINIDX;
        cv[slot] = -3.4e38f;
      } else {
        vals[(size_t)row * KTOP + k] = 0.f;
        idx[(size_t)row * KTOP + k]  = MINIDX;
      }
    }
    __syncthreads();
  }
}

__global__ void fill_zero_kernel(float* __restrict__ p, int n) {
  int i = blockIdx.x * blockDim.x + threadIdx.x;
  if (i < n) p[i] = 0.f;
}

// ---------------------------------------------------------------------------
extern "C" void kernel_launch(void* const* d_in, const int* in_sizes, int n_in,
                              void* d_out, int out_size, void* d_ws, size_t ws_size,
                              hipStream_t stream) {
  const float* user_emb = (const float*)d_in[0];
  const float* item_emb = (const float*)d_in[1];
  const int*   hrows    = (const int*)d_in[2];
  const int*   hcols    = (const int*)d_in[3];
  const float* W1  = (const float*)d_in[4];
  const float* b1  = (const float*)d_in[5];
  const float* g1  = (const float*)d_in[6];
  const float* be1 = (const float*)d_in[7];
  const float* W2  = (const float*)d_in[8];
  const float* b2  = (const float*)d_in[9];
  const float* g2  = (const float*)d_in[10];
  const float* be2 = (const float*)d_in[11];

  // workspace: scores f32 | encU bf16 | encI bf16 | W1T bf16 | W2T bf16
  char* ws = (char*)d_ws;
  float* S = (float*)ws;
  unsigned short* encU = (unsigned short*)(ws + 409600000ull);
  unsigned short* encI = (unsigned short*)(ws + 409600000ull + 524288ull);
  unsigned short* W1T  = (unsigned short*)(ws + 461324288ull);
  unsigned short* W2T  = (unsigned short*)(ws + 461324288ull + 131072ull);

  // d_out layout: topk_vals | topk_idx | probs | user_loss | item_loss | add_loss
  float* topv   = (float*)d_out;
  int*   topi   = (int*)d_out + (size_t)BDIM * KTOP;
  float* probs  = (float*)d_out + 2ull * BDIM * KTOP;
  float* losses = (float*)d_out + 2ull * BDIM * KTOP + (size_t)BDIM * NITEMS;

  // one-time bf16 transposed weights
  prep_wT_kernel<<<DIMD, DIMD, 0, stream>>>(W1, W1T);
  prep_wT_kernel<<<DIMD, DIMD, 0, stream>>>(W2, W2T);

  encode_kernel<<<BDIM / 16, 256, 0, stream>>>(user_emb, W1T, b1, g1, be1,
                                               W2T, b2, g2, be2, encU);
  // items = item_embeddings[MIN_IDX:]  -> skip first row
  encode_kernel<<<NITEMS / 16, 256, 0, stream>>>(item_emb + DIMD, W1T, b1, g1, be1,
                                                 W2T, b2, g2, be2, encI);

  dim3 gS((NITEMS + 255) / 256, BDIM / 16);
  scores_kernel<<<gS, 256, 0, stream>>>(encU, encI, S);

  zero_hist_kernel<<<(EHIST + 255) / 256, 256, 0, stream>>>(hrows, hcols, S, EHIST);

  const long long n4 = (long long)BDIM * NITEMS / 4;
  sigmoid_kernel<<<4096, 256, 0, stream>>>(S, probs, n4);

  topk_kernel<<<BDIM, 256, 0, stream>>>(S, topv, topi);

  fill_zero_kernel<<<(BDIM + NITEMS + 1 + 255) / 256, 256, 0, stream>>>(
      losses, BDIM + NITEMS + 1);
}
